// LSTMLayer_56100862820648
// MI455X (gfx1250) — matrix operand order repaired and checked by reading, not compiled, not run on hardware
//
#include <hip/hip_runtime.h>

// ---------------- types ----------------
typedef __attribute__((ext_vector_type(16))) __bf16 v16bf;
typedef __attribute__((ext_vector_type(8)))  float  v8f;
typedef __attribute__((ext_vector_type(4)))  int    i4;
typedef __attribute__((ext_vector_type(4)))  float  f4;
typedef __attribute__((ext_vector_type(4)))  __bf16 v4bf;

union FragBF { v16bf v; i4 q[2]; };

#define BATCH 4096
#define DIN   2048
#define HDIM  2048
#define KDIM  4096   // DIN + HDIM
#define TM    128
#define TN    32     // per-gate N tile
#define BK    64     // staged K per LDS buffer (2 WMMA k-substeps)

__device__ __forceinline__ float sigmoid_f(float x) {
    return 1.0f / (1.0f + __expf(-x));
}
__device__ __forceinline__ float tanh_f(float x) {
    return 1.0f - 2.0f / (1.0f + __expf(2.0f * x));
}

// ---------------- pack kernels (fp32 -> bf16) ----------------
__global__ __launch_bounds__(256) void lstm_pack_u(const float* __restrict__ in,
                                                   const float* __restrict__ hid,
                                                   __bf16* __restrict__ U) {
    size_t idx = ((size_t)blockIdx.x * blockDim.x + threadIdx.x) * 4;
    int b = (int)(idx >> 12);
    int k = (int)(idx & 4095);
    const float* src = (k < DIN) ? (in + (size_t)b * DIN + k)
                                 : (hid + (size_t)b * HDIM + (k - DIN));
    f4 x = *(const f4*)src;
    v4bf y;
    y[0] = (__bf16)x[0]; y[1] = (__bf16)x[1];
    y[2] = (__bf16)x[2]; y[3] = (__bf16)x[3];
    *(v4bf*)(U + idx) = y;
}

__global__ __launch_bounds__(256) void lstm_pack_w(const float* __restrict__ Wi,
                                                   const float* __restrict__ Wf,
                                                   const float* __restrict__ Wc,
                                                   const float* __restrict__ Wo,
                                                   __bf16* __restrict__ Wb) {
    size_t idx = ((size_t)blockIdx.x * blockDim.x + threadIdx.x) * 4;
    int n = (int)(idx >> 12);
    int k = (int)(idx & 4095);
    int g = n >> 11;
    int r = n & 2047;
    const float* Ws[4] = {Wi, Wf, Wc, Wo};
    const float* src = Ws[g] + (size_t)r * KDIM + k;
    f4 x = *(const f4*)src;
    v4bf y;
    y[0] = (__bf16)x[0]; y[1] = (__bf16)x[1];
    y[2] = (__bf16)x[2]; y[3] = (__bf16)x[3];
    *(v4bf*)(Wb + idx) = y;
}

// ---------------- fused GEMM + LSTM cell ----------------
// grid.x = HDIM/TN = 64 (N tiles), grid.y = BATCH/TM = 32 (M tiles), 256 thr.
__global__ __launch_bounds__(256) void lstm_gemm(const __bf16* __restrict__ U,
                                                 const __bf16* __restrict__ Wb,
                                                 const float* __restrict__ cell,
                                                 const float* __restrict__ b_i,
                                                 const float* __restrict__ b_f,
                                                 const float* __restrict__ b_c,
                                                 const float* __restrict__ b_o,
                                                 float* __restrict__ outH,
                                                 float* __restrict__ outC) {
    __shared__ __bf16 sA[2][TM * BK];       // 2 x 16 KB
    __shared__ __bf16 sB[2][4 * TN * BK];   // 2 x 16 KB

    const int tid    = threadIdx.x;
    const int lane   = tid & 31;
    const int wave   = tid >> 5;            // 0..7
    const int mgroup = wave >> 1;           // 0..3 -> 32-row slab
    const int ngroup = wave & 1;            // 0..1 -> 16-col slab
    const int tileN  = blockIdx.x * TN;
    const int tileM  = blockIdx.y * TM;

    // cooperative global->LDS mapping: 4 b128 chunks per thread per array
    const int ld_row = tid >> 1;                  // 0..127 (A row)
    const int ld_c0  = (tid & 1) * 8;             // chunk phase (elements)
    const int bg     = tid >> 6;                  // 0..3 (gate)
    const int b_row  = (tid >> 1) & 31;           // 0..31 (B row in tile)
    const __bf16* aSrc = U  + (size_t)(tileM + ld_row) * KDIM + ld_c0;
    const __bf16* bSrc = Wb + (size_t)(bg * HDIM + tileN + b_row) * KDIM + ld_c0;
    const int aDstOff = ld_row * BK + ld_c0;
    const int bDstOff = (bg * TN + b_row) * BK + ld_c0;

    // fragment addressing (ISA 7.12.2 layouts)
    const int half16 = lane >> 4;
    const int l15    = lane & 15;

    v8f acc[4][2];
    const v8f vzero = {0.f, 0.f, 0.f, 0.f, 0.f, 0.f, 0.f, 0.f};
#pragma unroll
    for (int g = 0; g < 4; ++g)
#pragma unroll
        for (int mi = 0; mi < 2; ++mi) acc[g][mi] = vzero;

    i4 stA[4], stB[4];
    const int NITER = KDIM / BK;                  // 64

    // prologue: tile 0 -> buffer 0
#pragma unroll
    for (int j = 0; j < 4; ++j) {
        stA[j] = *(const i4*)(aSrc + j * 16);
        stB[j] = *(const i4*)(bSrc + j * 16);
    }
#pragma unroll
    for (int j = 0; j < 4; ++j) {
        *(i4*)(&sA[0][aDstOff + j * 16]) = stA[j];
        *(i4*)(&sB[0][bDstOff + j * 16]) = stB[j];
    }
    __syncthreads();
    // stage tile 1 into registers
#pragma unroll
    for (int j = 0; j < 4; ++j) {
        stA[j] = *(const i4*)(aSrc + BK + j * 16);
        stB[j] = *(const i4*)(bSrc + BK + j * 16);
    }

    for (int it = 0; it < NITER; ++it) {
        const int cur = it & 1;
        const int nxt = cur ^ 1;

        // park tile it+1 in the alternate buffer (its readers finished at
        // the barrier ending iteration it-1)
        if (it + 1 < NITER) {
#pragma unroll
            for (int j = 0; j < 4; ++j) {
                *(i4*)(&sA[nxt][aDstOff + j * 16]) = stA[j];
                *(i4*)(&sB[nxt][bDstOff + j * 16]) = stB[j];
            }
        }
        // issue global loads for tile it+2 (clamped; latency hidden by WMMAs)
        const int k2 = (it + 2 < NITER) ? (it + 2) * BK : (NITER - 1) * BK;
        __builtin_prefetch(aSrc + k2 + BK, 0, 1);   // global_prefetch_b8
        __builtin_prefetch(bSrc + k2 + BK, 0, 1);
#pragma unroll
        for (int j = 0; j < 4; ++j) {
            stA[j] = *(const i4*)(aSrc + k2 + j * 16);
            stB[j] = *(const i4*)(bSrc + k2 + j * 16);
        }

        // compute tile it: two 32-wide K substeps, 16 WMMAs per wave
#pragma unroll
        for (int ks = 0; ks < BK; ks += 32) {
            FragBF a[2];
#pragma unroll
            for (int mi = 0; mi < 2; ++mi) {
                const __bf16* p =
                    &sA[cur][(mgroup * 32 + mi * 16 + l15) * BK + ks + half16 * 8];
                a[mi].q[0] = *(const i4*)p;
                a[mi].q[1] = *(const i4*)(p + 16);
            }
            FragBF b[4];
#pragma unroll
            for (int g = 0; g < 4; ++g) {
                const __bf16* p =
                    &sB[cur][(g * TN + ngroup * 16 + l15) * BK + ks + half16 * 8];
                b[g].q[0] = *(const i4*)p;
                b[g].q[1] = *(const i4*)(p + 16);
            }
#pragma unroll
            for (int g = 0; g < 4; ++g)
#pragma unroll
                for (int mi = 0; mi < 2; ++mi)
                    acc[g][mi] = __builtin_amdgcn_wmma_f32_16x16x32_bf16(
                        false, a[mi].v, false, b[g].v, (short)0, acc[g][mi],
                        false, false);
        }
        __syncthreads();   // single barrier per staged tile
    }

    // -------- epilogue: bias + activations + cell update, all in regs --------
    const int nH = tileN + ngroup * 16 + l15;
    const float biV = b_i[nH];
    const float bfV = b_f[nH];
    const float bcV = b_c[nH];
    const float boV = b_o[nH];

#pragma unroll
    for (int mi = 0; mi < 2; ++mi) {
        const int mBase = tileM + mgroup * 32 + mi * 16 + half16 * 8;
#pragma unroll
        for (int r = 0; r < 8; ++r) {
            const int m = mBase + r;
            const size_t o = (size_t)m * HDIM + nH;
            float gi = acc[0][mi][r] + biV;
            float gf = acc[1][mi][r] + bfV;
            float gc = acc[2][mi][r] + bcV;
            float go = acc[3][mi][r] + boV;
            float ig = sigmoid_f(gi);
            float fg = sigmoid_f(gf);
            float nc = tanh_f(gc);
            float og = sigmoid_f(go);
            float c_new = cell[o] * fg + ig * nc;
            outC[o] = c_new;
            outH[o] = og * tanh_f(c_new);
        }
    }
}

// ---------------- launch ----------------
extern "C" void kernel_launch(void* const* d_in, const int* in_sizes, int n_in,
                              void* d_out, int out_size, void* d_ws, size_t ws_size,
                              hipStream_t stream) {
    const float* input  = (const float*)d_in[0];
    const float* hidden = (const float*)d_in[1];
    const float* cellst = (const float*)d_in[2];
    const float* Wi     = (const float*)d_in[3];
    const float* bi     = (const float*)d_in[4];
    const float* Wf     = (const float*)d_in[5];
    const float* bf     = (const float*)d_in[6];
    const float* Wc     = (const float*)d_in[7];
    const float* bc     = (const float*)d_in[8];
    const float* Wo     = (const float*)d_in[9];
    const float* bo     = (const float*)d_in[10];

    float* outH = (float*)d_out;
    float* outC = outH + (size_t)BATCH * HDIM;

    __bf16* U  = (__bf16*)d_ws;                   // 4096*4096 bf16 = 32 MB
    __bf16* Wb = U + (size_t)BATCH * KDIM;        // 8192*4096 bf16 = 64 MB

    lstm_pack_u<<<(BATCH * (size_t)KDIM) / (4 * 256), 256, 0, stream>>>(input, hidden, U);
    lstm_pack_w<<<(4 * (size_t)HDIM * KDIM) / (4 * 256), 256, 0, stream>>>(Wi, Wf, Wc, Wo, Wb);

    dim3 grid(HDIM / TN, BATCH / TM);             // 64 x 32
    lstm_gemm<<<grid, 256, 0, stream>>>(U, Wb, cellst, bi, bf, bc, bo, outH, outC);
}
